// ESPERNetEncoder_72619307041313
// MI455X (gfx1250) — matrix operand order repaired and checked by reading, not compiled, not run on hardware
//
#include <hip/hip_runtime.h>

// ---------------- types ----------------
typedef __bf16 bf16;
typedef __attribute__((ext_vector_type(16))) __bf16 v16bf;
typedef __attribute__((ext_vector_type(8)))  float  v8f;

// ---------------- model constants ----------------
#define BB    16
#define TT    1024
#define SS    1025          // T + cls
#define DD    512
#define HH    8
#define HDD   64
#define FFD   2048
#define NLAY  6
#define KPRE  320           // 290 feats + 8 pitch + 8 pos, padded 306 -> 320
#define MROW  (BB*SS)       // 16400
#define MTOK  (BB*TT)       // 16384

// low 32 bits of a flat pointer to LDS == wave-relative LDS byte offset
// (async ops add the instruction offset to the LDS address too)
__device__ __forceinline__ unsigned lds_off(const void* p)
{
    return (unsigned)(uintptr_t)p;
}

// =====================================================================
// WMMA GEMM:  C[M,N] = A[M,K](bf16) * W[N,K](bf16)^T + bias[N]
// block tile 128x128, 256 threads = 8 waves (4 M-waves x 2 N-waves),
// wave tile 32x64 = 2x4 fragments of v_wmma_f32_16x16x32_bf16.
// Tiles staged with GLOBAL_LOAD_ASYNC_TO_LDS_B128 into double-buffered LDS,
// pipelined with s_wait_asynccnt (4 async ops/wave/tile, in-order per wave).
// Requires: K % 32 == 0, N % 128 == 0. M guarded.
// =====================================================================
template<typename OutT, bool RELU>
__global__ __launch_bounds__(256)
void gemm_bf16_wmma(const bf16* __restrict__ A, const bf16* __restrict__ W,
                    const float* __restrict__ bias, OutT* __restrict__ C,
                    int M, int N, int K)
{
    constexpr int LDT = 40;                       // LDS row stride (bf16), 80B (16B-aligned)
    __shared__ bf16 Asm[2][128 * LDT];
    __shared__ bf16 Wsm[2][128 * LDT];

    const int tid  = threadIdx.x;
    const int lane = tid & 31;
    const int wave = tid >> 5;
    const int wm   = wave >> 1;                   // 0..3
    const int wn   = wave & 1;                    // 0..1
    const int tileM = blockIdx.x * 128;
    const int tileN = blockIdx.y * 128;

    v8f acc[2][4];
#pragma unroll
    for (int f = 0; f < 2; ++f)
#pragma unroll
        for (int g = 0; g < 4; ++g)
#pragma unroll
            for (int r = 0; r < 8; ++r) acc[f][g][r] = 0.f;

    // ---- async tile staging: 256 threads, 2 per row, 32B each (2 x b128) ----
    const int lr = tid >> 1;                      // 0..127
    const int lc = (tid & 1) * 16;                // 0 or 16 (bf16 elems)
    const bool abad = (tileM + lr) >= M;          // only in last M-tile
    const int  arow = abad ? tileM : (tileM + lr);
    const bf16* Agp = A + (size_t)arow * K + lc;
    const bf16* Wgp = W + (size_t)(tileN + lr) * K + lc;   // N % 128 == 0, always in-bounds
    unsigned ldsAo[2], ldsWo[2];
#pragma unroll
    for (int s = 0; s < 2; ++s) {
        ldsAo[s] = lds_off(&Asm[s][lr * LDT + lc]);
        ldsWo[s] = lds_off(&Wsm[s][lr * LDT + lc]);
    }

    auto stage = [&](int kk, int bsel) {
        const bf16* ga = Agp + kk;
        const bf16* gw = Wgp + kk;
        asm volatile("global_load_async_to_lds_b128 %0, %1, off"
                     :: "v"(ldsAo[bsel]), "v"(ga) : "memory");
        asm volatile("global_load_async_to_lds_b128 %0, %1, off offset:16"
                     :: "v"(ldsAo[bsel]), "v"(ga) : "memory");
        asm volatile("global_load_async_to_lds_b128 %0, %1, off"
                     :: "v"(ldsWo[bsel]), "v"(gw) : "memory");
        asm volatile("global_load_async_to_lds_b128 %0, %1, off offset:16"
                     :: "v"(ldsWo[bsel]), "v"(gw) : "memory");
    };

    const int l16 = lane & 15;
    const int kg  = lane >> 4;                    // K-group select
    const int nkt = K >> 5;                       // # of 32-wide K tiles

    stage(0, 0);
    for (int i = 0; i < nkt; ++i) {
        const int cb = i & 1;
        if (i + 1 < nkt) {
            stage((i + 1) << 5, cb ^ 1);
            // current tile's 4 asyncs done once count drops to the 4 prefetch ops
            asm volatile("s_wait_asynccnt 4" ::: "memory");
        } else {
            asm volatile("s_wait_asynccnt 0" ::: "memory");
        }
        if (abad) {                               // zero-pad out-of-range A rows
            uint4 z{0, 0, 0, 0};
            uint4* s = (uint4*)&Asm[cb][lr * LDT + lc];
            s[0] = z; s[1] = z;
        }
        __syncthreads();

        const unsigned* As32 = (const unsigned*)Asm[cb];
        const unsigned* Ws32 = (const unsigned*)Wsm[cb];

        // A fragments (ISA 16-bit A 16x32 layout: lanes 0-15 K=0..7,16..23; lanes 16-31 K=8..15,24..31)
        union Frag { v16bf v; unsigned u[8]; };
        Frag a[2], b[4];
#pragma unroll
        for (int f = 0; f < 2; ++f) {
            const int row = wm * 32 + f * 16 + l16;
            const unsigned* base = As32 + row * (LDT / 2);
#pragma unroll
            for (int j = 0; j < 4; ++j) {
                a[f].u[j]     = base[kg * 4 + j];         // K = kg*8 + 2j
                a[f].u[4 + j] = base[kg * 4 + 8 + j];     // K = kg*8 + 16 + 2j
            }
        }
        // B fragments (32x16: lanes 0-15 N, K=0..7,16..23; lanes 16-31 K=8..15,24..31)
#pragma unroll
        for (int g = 0; g < 4; ++g) {
            const int nrow = wn * 64 + g * 16 + l16;
            const unsigned* base = Ws32 + nrow * (LDT / 2);
#pragma unroll
            for (int j = 0; j < 4; ++j) { b[g].u[j] = base[kg * 4 + j]; b[g].u[4 + j] = base[kg * 4 + 8 + j]; }
        }
#pragma unroll
        for (int f = 0; f < 2; ++f)
#pragma unroll
            for (int g = 0; g < 4; ++g)
                acc[f][g] = __builtin_amdgcn_wmma_f32_16x16x32_bf16(
                    false, a[f].v, false, b[g].v, (short)0, acc[f][g], false, false);
        __syncthreads();
    }

    // epilogue: C/D layout -> VGPR r: M = (lane>>4)*8 + r, N = lane&15
#pragma unroll
    for (int f = 0; f < 2; ++f) {
#pragma unroll
        for (int g = 0; g < 4; ++g) {
            const int col = tileN + wn * 64 + g * 16 + l16;
            const float bv = bias[col];
#pragma unroll
            for (int r = 0; r < 8; ++r) {
                const int row = tileM + wm * 32 + f * 16 + kg * 8 + r;
                if (row < M) {
                    float v = acc[f][g][r] + bv;
                    if (RELU) v = v > 0.f ? v : 0.f;
                    C[(size_t)row * N + col] = (OutT)v;
                }
            }
        }
    }
}

// =====================================================================
// feature prep: feats[b,t, 0..319] = [x[...,1:291], sincos(pitch,8), sincos(t,8), pad]
// =====================================================================
__global__ __launch_bounds__(320)
void prep_feats(const float* __restrict__ x, bf16* __restrict__ feats)
{
    const int t = blockIdx.x, b = blockIdx.y, c = threadIdx.x;
    const float* xr = x + ((size_t)b * TT + t) * 291;
    float v = 0.f;
    if (c < 290) {
        v = xr[1 + c];
    } else if (c < 298) {
        const int i = c - 290;
        const int fi = (i < 4) ? i : i - 4;
        const float freq = powf(10000.f, -(float)fi / 4.f);
        const float ang = xr[0] * freq;
        v = (i < 4) ? sinf(ang) : cosf(ang);
    } else if (c < 306) {
        const int i = c - 298;
        const int fi = (i < 4) ? i : i - 4;
        const float freq = powf(10000.f, -(float)fi / 4.f);
        const float ang = (float)t * freq;
        v = (i < 4) ? sinf(ang) : cosf(ang);
    }
    feats[((size_t)b * TT + t) * KPRE + c] = (bf16)v;
}

// float -> bf16 elementwise
__global__ void f2bf(const float* __restrict__ in, bf16* __restrict__ out, int n)
{
    const int i = blockIdx.x * 256 + threadIdx.x;
    if (i < n) out[i] = (bf16)in[i];
}

// pre_W [512,306] -> bf16 padded [512,320]
__global__ void conv_preW(const float* __restrict__ in, bf16* __restrict__ out)
{
    const int i = blockIdx.x * 256 + threadIdx.x;
    if (i >= DD * KPRE) return;
    const int n = i / KPRE, k = i % KPRE;
    out[i] = (bf16)((k < 306) ? in[n * 306 + k] : 0.f);
}

// assemble h[B,S,D]: rows t<1024 from pre-GEMM result (bias already added), row 1024 = cls
__global__ void assemble_h(const float* __restrict__ pre, const float* __restrict__ cls,
                           float* __restrict__ h, bf16* __restrict__ hb)
{
    const int idx = blockIdx.x * 256 + threadIdx.x;
    if (idx >= MROW * DD) return;
    const int row = idx >> 9, d = idx & 511;
    const int b = row / SS, s = row % SS;
    const float v = (s < TT) ? pre[((size_t)b * TT + s) * DD + d] : cls[d];
    h[idx] = v;
    hb[idx] = (bf16)v;
}

// =====================================================================
// windowed attention (|q-k| <= 3), one wave per (b,h,q), 2 dims/lane.
// qkv bf16 [MROW, 3*D] (q | k | v), out bf16 [MROW, D]
// =====================================================================
__global__ __launch_bounds__(256)
void attn_window(const bf16* __restrict__ qkv, bf16* __restrict__ out)
{
    const int lane = threadIdx.x & 31;
    const int wave = threadIdx.x >> 5;
    const int s = blockIdx.x * 8 + wave;          // 0..1023
    const int hh = blockIdx.y, b = blockIdx.z;

    const size_t qrow = ((size_t)b * SS + s) * (3 * DD) + hh * HDD;
    const float q0 = (float)qkv[qrow + lane];
    const float q1 = (float)qkv[qrow + lane + 32];

    int klo = s - 3; if (klo < 0) klo = 0;
    int khi = s + 3; if (khi > SS - 1) khi = SS - 1;
    const int nk = khi - klo + 1;

    float sc[7];
    for (int j = 0; j < nk; ++j) {
        const size_t krow = ((size_t)b * SS + klo + j) * (3 * DD) + DD + hh * HDD;
        float p = q0 * (float)qkv[krow + lane] + q1 * (float)qkv[krow + lane + 32];
        for (int off = 16; off; off >>= 1) p += __shfl_xor(p, off);
        sc[j] = p * 0.125f;                        // 1/sqrt(64)
    }
    float mx = -1e30f;
    for (int j = 0; j < nk; ++j) mx = fmaxf(mx, sc[j]);
    float sum = 0.f;
    for (int j = 0; j < nk; ++j) { sc[j] = expf(sc[j] - mx); sum += sc[j]; }
    const float inv = 1.f / sum;

    float o0 = 0.f, o1 = 0.f;
    for (int j = 0; j < nk; ++j) {
        const size_t vrow = ((size_t)b * SS + klo + j) * (3 * DD) + 2 * DD + hh * HDD;
        const float w = sc[j] * inv;
        o0 += w * (float)qkv[vrow + lane];
        o1 += w * (float)qkv[vrow + lane + 32];
    }
    const size_t orow = ((size_t)b * SS + s) * DD + hh * HDD;
    out[orow + lane]      = (bf16)o0;
    out[orow + lane + 32] = (bf16)o1;
}

// CLS row: full attention over all 1025 keys. one block per (b,h).
__global__ __launch_bounds__(256)
void attn_cls(const bf16* __restrict__ qkv, bf16* __restrict__ out)
{
    __shared__ float sc[SS];
    __shared__ float red[8];
    __shared__ float qv[HDD];
    const int hh = blockIdx.x, b = blockIdx.y;
    const int tid = threadIdx.x;

    const size_t qrow = ((size_t)b * SS + (SS - 1)) * (3 * DD) + hh * HDD;
    if (tid < HDD) qv[tid] = (float)qkv[qrow + tid];
    __syncthreads();

    float lmax = -1e30f;
    for (int j = tid; j < SS; j += 256) {
        const size_t krow = ((size_t)b * SS + j) * (3 * DD) + DD + hh * HDD;
        float p = 0.f;
        for (int d = 0; d < HDD; ++d) p += qv[d] * (float)qkv[krow + d];
        p *= 0.125f;
        sc[j] = p;
        lmax = fmaxf(lmax, p);
    }
    for (int off = 16; off; off >>= 1) lmax = fmaxf(lmax, __shfl_xor(lmax, off));
    if ((tid & 31) == 0) red[tid >> 5] = lmax;
    __syncthreads();
    float bmax = red[0];
    for (int w = 1; w < 8; ++w) bmax = fmaxf(bmax, red[w]);

    float lsum = 0.f;
    for (int j = tid; j < SS; j += 256) { const float e = expf(sc[j] - bmax); sc[j] = e; lsum += e; }
    for (int off = 16; off; off >>= 1) lsum += __shfl_xor(lsum, off);
    __syncthreads();
    if ((tid & 31) == 0) red[tid >> 5] = lsum;
    __syncthreads();
    float bsum = 0.f;
    for (int w = 0; w < 8; ++w) bsum += red[w];
    const float inv = 1.f / bsum;

    if (tid < HDD) {
        float o = 0.f;
        for (int j = 0; j < SS; ++j) {
            const size_t vrow = ((size_t)b * SS + j) * (3 * DD) + 2 * DD + hh * HDD;
            o += sc[j] * (float)qkv[vrow + tid];
        }
        out[((size_t)b * SS + (SS - 1)) * DD + hh * HDD + tid] = (bf16)(o * inv);
    }
}

// =====================================================================
// fused residual add + LayerNorm; writes f32 h and bf16 mirror. 1 block / row.
// =====================================================================
__global__ __launch_bounds__(256)
void add_ln(const float* __restrict__ hin, const float* __restrict__ delta,
            const float* __restrict__ g, const float* __restrict__ be,
            float* __restrict__ hout, bf16* __restrict__ hb)
{
    __shared__ float redS[8], redQ[8], stat[2];
    const int row = blockIdx.x, tid = threadIdx.x;
    const size_t base = (size_t)row * DD;

    const float x0 = hin[base + tid]       + delta[base + tid];
    const float x1 = hin[base + tid + 256] + delta[base + tid + 256];
    float s = x0 + x1, s2 = x0 * x0 + x1 * x1;
    for (int off = 16; off; off >>= 1) { s += __shfl_xor(s, off); s2 += __shfl_xor(s2, off); }
    if ((tid & 31) == 0) { redS[tid >> 5] = s; redQ[tid >> 5] = s2; }
    __syncthreads();
    if (tid == 0) {
        float ts = 0.f, tq = 0.f;
        for (int w = 0; w < 8; ++w) { ts += redS[w]; tq += redQ[w]; }
        const float mean = ts / (float)DD;
        const float var  = tq / (float)DD - mean * mean;
        stat[0] = mean; stat[1] = rsqrtf(var + 1e-5f);
    }
    __syncthreads();
    const float mean = stat[0], rstd = stat[1];
    const float y0 = (x0 - mean) * rstd * g[tid]       + be[tid];
    const float y1 = (x1 - mean) * rstd * g[tid + 256] + be[tid + 256];
    hout[base + tid] = y0;       hout[base + tid + 256] = y1;
    hb[base + tid]   = (bf16)y0; hb[base + tid + 256]   = (bf16)y1;
}

// ---------------- heads ----------------
__global__ __launch_bounds__(128)
void voice_head(const float* __restrict__ h, const float* __restrict__ W,
                const float* __restrict__ bias, float* __restrict__ out)
{
    const int b = blockIdx.x, n = threadIdx.x;
    const float* hr = h + ((size_t)b * SS + (SS - 1)) * DD;
    const float* wr = W + (size_t)n * DD;
    float acc = bias[n];
    for (int d = 0; d < DD; ++d) acc += hr[d] * wr[d];
    out[b * 128 + n] = acc;
}

__global__ void pitch_copy(const float* __restrict__ x, float* __restrict__ out)
{
    const int i = blockIdx.x * 256 + threadIdx.x;
    if (i < MTOK) out[i] = x[(size_t)i * 291];
}

__global__ void phon_head(const float* __restrict__ h, const float* __restrict__ W,
                          const float* __restrict__ bias, float* __restrict__ out)
{
    const int idx = blockIdx.x * 256 + threadIdx.x;
    if (idx >= MTOK * 10) return;
    const int row = idx / 10, n = idx % 10;
    const int b = row >> 10, t = row & 1023;
    const float* hr = h + ((size_t)b * SS + t) * DD;
    const float* wr = W + n * DD;
    float acc = bias[n];
    for (int d = 0; d < DD; ++d) acc += hr[d] * wr[d];
    out[idx] = acc;
}

// =====================================================================
// launcher
// =====================================================================
extern "C" void kernel_launch(void* const* d_in, const int* in_sizes, int n_in,
                              void* d_out, int out_size, void* d_ws, size_t ws_size,
                              hipStream_t stream)
{
    const float* x        = (const float*)d_in[0];
    const float* cls      = (const float*)d_in[1];
    const float* pre_W    = (const float*)d_in[2];
    const float* pre_b    = (const float*)d_in[3];
    const float* inproj_W = (const float*)d_in[4];
    const float* inproj_b = (const float*)d_in[5];
    const float* out_W    = (const float*)d_in[6];
    const float* out_b    = (const float*)d_in[7];
    const float* ff1_W    = (const float*)d_in[8];
    const float* ff1_b    = (const float*)d_in[9];
    const float* ff2_W    = (const float*)d_in[10];
    const float* ff2_b    = (const float*)d_in[11];
    const float* ln1_w    = (const float*)d_in[12];
    const float* ln1_b    = (const float*)d_in[13];
    const float* ln2_w    = (const float*)d_in[14];
    const float* ln2_b    = (const float*)d_in[15];
    const float* voice_W  = (const float*)d_in[16];
    const float* voice_b  = (const float*)d_in[17];
    const float* phon_W   = (const float*)d_in[18];
    const float* phon_b   = (const float*)d_in[19];
    float* out = (float*)d_out;

    // ---- workspace layout (bump allocator, 256B aligned; ~189 MB total) ----
    char* ws = (char*)d_ws;
    size_t off = 0;
    auto alloc = [&](size_t bytes) { char* p = ws + off; off += (bytes + 255) & ~(size_t)255; return p; };
    bf16*  preWb  = (bf16*)alloc((size_t)DD * KPRE * 2);
    bf16*  Wqkvb  = (bf16*)alloc((size_t)NLAY * 3 * DD * DD * 2);
    bf16*  Wob    = (bf16*)alloc((size_t)NLAY * DD * DD * 2);
    bf16*  W1b    = (bf16*)alloc((size_t)NLAY * FFD * DD * 2);
    bf16*  W2b    = (bf16*)alloc((size_t)NLAY * DD * FFD * 2);
    float* h      = (float*)alloc((size_t)MROW * DD * 4);
    bf16*  hb     = (bf16*)alloc((size_t)MROW * DD * 2);
    bf16*  buf1   = (bf16*)alloc((size_t)MROW * FFD * 2);   // qkv bf16 / ff1 bf16 / feats
    float* buf2   = (float*)alloc((size_t)MROW * DD * 4);   // f32 GEMM outs

    // ---- convert weights to bf16 (deterministic every call) ----
    conv_preW<<<(DD * KPRE + 255) / 256, 256, 0, stream>>>(pre_W, preWb);
    {
        int n;
        n = NLAY * 3 * DD * DD; f2bf<<<(n + 255) / 256, 256, 0, stream>>>(inproj_W, Wqkvb, n);
        n = NLAY * DD * DD;     f2bf<<<(n + 255) / 256, 256, 0, stream>>>(out_W, Wob, n);
        n = NLAY * FFD * DD;    f2bf<<<(n + 255) / 256, 256, 0, stream>>>(ff1_W, W1b, n);
        n = NLAY * DD * FFD;    f2bf<<<(n + 255) / 256, 256, 0, stream>>>(ff2_W, W2b, n);
    }

    // ---- pre-projection ----
    prep_feats<<<dim3(TT, BB), 320, 0, stream>>>(x, buf1);
    gemm_bf16_wmma<float, false><<<dim3(MTOK / 128, DD / 128), 256, 0, stream>>>(
        buf1, preWb, pre_b, buf2, MTOK, DD, KPRE);
    assemble_h<<<(MROW * DD + 255) / 256, 256, 0, stream>>>(buf2, cls, h, hb);

    // ---- transformer layers ----
    for (int l = 0; l < NLAY; ++l) {
        const bf16* Wqkv = Wqkvb + (size_t)l * 3 * DD * DD;
        const bf16* Wo   = Wob   + (size_t)l * DD * DD;
        const bf16* W1   = W1b   + (size_t)l * FFD * DD;
        const bf16* W2   = W2b   + (size_t)l * DD * FFD;

        // qkv = hb @ Wqkv^T + b  -> bf16
        gemm_bf16_wmma<bf16, false><<<dim3((MROW + 127) / 128, (3 * DD) / 128), 256, 0, stream>>>(
            hb, Wqkv, inproj_b + l * 3 * DD, buf1, MROW, 3 * DD, DD);
        // attention -> hb (bf16)
        attn_window<<<dim3(TT / 8, HH, BB), 256, 0, stream>>>(buf1, hb);
        attn_cls<<<dim3(HH, BB), 256, 0, stream>>>(buf1, hb);
        // out proj -> f32
        gemm_bf16_wmma<float, false><<<dim3((MROW + 127) / 128, DD / 128), 256, 0, stream>>>(
            hb, Wo, out_b + l * DD, buf2, MROW, DD, DD);
        // h = LN(h + proj)
        add_ln<<<MROW, 256, 0, stream>>>(h, buf2, ln1_w + l * DD, ln1_b + l * DD, h, hb);
        // ff1 = relu(hb @ W1^T + b1) -> bf16
        gemm_bf16_wmma<bf16, true><<<dim3((MROW + 127) / 128, FFD / 128), 256, 0, stream>>>(
            hb, W1, ff1_b + l * FFD, buf1, MROW, FFD, DD);
        // ff2 -> f32
        gemm_bf16_wmma<float, false><<<dim3((MROW + 127) / 128, DD / 128), 256, 0, stream>>>(
            buf1, W2, ff2_b + l * DD, buf2, MROW, DD, FFD);
        // h = LN(h + ff)
        add_ln<<<MROW, 256, 0, stream>>>(h, buf2, ln2_w + l * DD, ln2_b + l * DD, h, hb);
    }

    // ---- heads: out = [voice(16*128) | pitch(16384) | phoneme(163840)] ----
    voice_head<<<BB, 128, 0, stream>>>(h, voice_W, voice_b, out);
    pitch_copy<<<(MTOK + 255) / 256, 256, 0, stream>>>(x, out + BB * 2 * 64);
    phon_head<<<(MTOK * 10 + 255) / 256, 256, 0, stream>>>(
        h, phon_W, phon_b, out + BB * 2 * 64 + MTOK);
}
